// DiffAttention_26731876451172
// MI455X (gfx1250) — matrix operand order repaired and checked by reading, not compiled, not run on hardware
//
#include <hip/hip_runtime.h>
#include <stdint.h>

// ---------------- problem constants ----------------
#define B_   2
#define L_   2048
#define H_   8
#define D_   64        // per-stream head dim
#define VD_  128       // value head dim (2*D)
#define DM_  1024      // model dim = H*2*D
#define BM   64        // query rows per workgroup
#define BN   64        // keys per LDS stage
#define NW   4         // waves per workgroup (wave32)
#define LDP  72        // padded LDS row stride (bf16 elems): 144B, 16B aligned, conflict-free
#define LAMBDA_INIT 0.8f
#define EPS_  1e-5f
#define NEGV (-1e9f)

// ---------------- CDNA5 WMMA types ----------------
typedef __attribute__((ext_vector_type(16))) __bf16 bf16x16;
typedef __attribute__((ext_vector_type(8)))  float  f32x8;
typedef __attribute__((ext_vector_type(4)))  unsigned int u32x4;
typedef __attribute__((ext_vector_type(4)))  float  f32x4;

union Frag16 {            // A/B operand: 16 bf16 = 8 VGPRs
    bf16x16        v;
    unsigned short u[16];
    u32x4          q[2];  // two 16-byte chunks for ds_load_b128
};
union Frag32 {            // C/D accumulator: 8 f32 = 8 VGPRs
    f32x8 v;
    float f[8];
};

__device__ __forceinline__ unsigned short f2bf(float x) {
    // round-to-nearest-even fp32 -> bf16
    unsigned int u = __float_as_uint(x);
    unsigned int r = u + 0x7FFFu + ((u >> 16) & 1u);
    return (unsigned short)(r >> 16);
}

// ---------------- CDNA5 async global->LDS copy (ASYNCcnt path) ----------------
#if __has_builtin(__builtin_amdgcn_global_load_async_to_lds_b128)
#define HAVE_ASYNC_LDS 1
#endif

// Builtin's parameter types (from the toolchain's own diagnostic): pointers to
// int __attribute__((vector_size(16))) in the global / LDS address spaces.
typedef int v4i_ __attribute__((vector_size(16)));
typedef __attribute__((address_space(1))) v4i_* as1_v4i_p;  // global (target AS 1)
typedef __attribute__((address_space(3))) v4i_* as3_v4i_p;  // LDS    (target AS 3)

__device__ __forceinline__ void async_copy_b128(const float* g, void* l) {
#ifdef HAVE_ASYNC_LDS
    // Integer-routed casts: generic LDS pointers keep the local byte offset in
    // their low 32 bits (flat aperture rule), global flat == AS1 numerically.
    __builtin_amdgcn_global_load_async_to_lds_b128(
        (as1_v4i_p)(uintptr_t)g,
        (as3_v4i_p)(unsigned)(uintptr_t)l,
        0, 0);
#else
    *(u32x4*)l = *(const u32x4*)g;   // synchronous fallback
#endif
}

__device__ __forceinline__ void wait_async_all() {
#ifdef HAVE_ASYNC_LDS
#if __has_builtin(__builtin_amdgcn_s_wait_asynccnt)
    __builtin_amdgcn_s_wait_asynccnt(0);
#else
    asm volatile("s_wait_asynccnt 0x0" ::: "memory");
#endif
#endif
}

// ---------------- dynamic LDS partition (bytes) ----------------
// bf16 operand tiles + fp32 async staging; ~118 KB of the 320 KB/WGP budget.
#define OFF_SK  0                         // ushort [2][BN][LDP]      = 18432
#define OFF_SVT (OFF_SK  + 2*BN*LDP*2)    // ushort [VD_][LDP]        = 18432
#define OFF_SP  (OFF_SVT + VD_*LDP*2)     // ushort [NW][2][16][LDP]  = 18432
#define OFF_FK  (OFF_SP  + NW*2*16*LDP*2) // float  [2][BN][D_]       = 32768
#define OFF_FV  (OFF_FK  + 2*BN*D_*4)     // float  [BN][VD_]         = 32768
#define SMEM_BYTES (OFF_FV + BN*VD_*4)    // 120832

extern "C" __global__ __launch_bounds__(NW * 32, 1)
void diffattn_fa_kernel(const float* __restrict__ q_g, const float* __restrict__ k_g,
                        const float* __restrict__ v_g,
                        const float* __restrict__ lq1, const float* __restrict__ lk1,
                        const float* __restrict__ lq2, const float* __restrict__ lk2,
                        const float* __restrict__ w_g, float* __restrict__ out) {
    extern __shared__ __align__(16) char smem[];
    unsigned short* sK  = (unsigned short*)(smem + OFF_SK);   // [s][key][feat]
    unsigned short* sVt = (unsigned short*)(smem + OFF_SVT);  // [ch][key]
    unsigned short* sP  = (unsigned short*)(smem + OFF_SP);   // [wave][s][row][col]
    float*          fK  = (float*)(smem + OFF_FK);            // fp32 staging [s][key][feat]
    float*          fV  = (float*)(smem + OFF_FV);            // fp32 staging [key][ch]

#define SK(s,k,f)      sK[(((s)*BN) + (k))*LDP + (f)]
#define SVT(c,k)       sVt[(c)*LDP + (k)]
#define SP(w,s,r,cc)   sP[((((w)*2 + (s))*16) + (r))*LDP + (cc)]
#define FK(s,k,f)      fK[(((s)*BN) + (k))*D_ + (f)]
#define FV(k,c)        fV[(k)*VD_ + (c)]

    const int qb   = blockIdx.x;       // query block (0..31)
    const int bh   = blockIdx.y;       // fused batch*head (0..15)
    const int b    = bh / H_;
    const int h    = bh % H_;
    const int tid  = threadIdx.x;
    const int wave = tid >> 5;
    const int lane = tid & 31;
    const int li   = lane & 15;        // N / column lane within half
    const int hi   = lane >> 4;        // which 16-lane half

    // ---- issue one key-block's worth of async global->LDS fp32 staging ----
    auto stage_async = [&](int kb) {
        // K, both streams: 2*64*64 fp32 = 2048 x 16B chunks, 16 per thread
#pragma unroll
        for (int it = 0; it < 16; ++it) {
            const int chunk = tid + it * 128;      // 0..2047
            const int s     = chunk >> 10;
            const int c     = chunk & 1023;
            const int key   = c >> 4;
            const int fc    = (c & 15) * 4;
            async_copy_b128(k_g + ((size_t)b * L_ + kb * BN + key) * DM_
                                + h * (2 * D_) + s * D_ + fc,
                            &FK(s, key, fc));
        }
        // V: 64*128 fp32 = 2048 x 16B chunks, 16 per thread
#pragma unroll
        for (int it = 0; it < 16; ++it) {
            const int chunk = tid + it * 128;      // 0..2047
            const int key   = chunk >> 5;
            const int cc    = (chunk & 31) * 4;
            async_copy_b128(v_g + ((size_t)b * L_ + kb * BN + key) * DM_ + h * VD_ + cc,
                            &FV(key, cc));
        }
    };

    // ---- LDS fp32 staging -> bf16 operand tiles (K natural, V transposed) ----
    auto convert_tiles = [&]() {
#pragma unroll
        for (int it = 0; it < 16; ++it) {
            const int chunk = tid + it * 128;
            const int s     = chunk >> 10;
            const int c     = chunk & 1023;
            const int key   = c >> 4;
            const int fc    = (c & 15) * 4;
            const f32x4 x = *(const f32x4*)&FK(s, key, fc);
            unsigned short* dst = &SK(s, key, fc);
            dst[0] = f2bf(x.x); dst[1] = f2bf(x.y); dst[2] = f2bf(x.z); dst[3] = f2bf(x.w);
        }
#pragma unroll
        for (int it = 0; it < 16; ++it) {
            const int chunk = tid + it * 128;
            const int key   = chunk >> 5;
            const int cc    = (chunk & 31) * 4;
            const f32x4 x = *(const f32x4*)&FV(key, cc);
            SVT(cc + 0, key) = f2bf(x.x); SVT(cc + 1, key) = f2bf(x.y);
            SVT(cc + 2, key) = f2bf(x.z); SVT(cc + 3, key) = f2bf(x.w);
        }
    };

    // ---- lambda scalar (tiny, redundantly computed per thread; L2-resident) ----
    float a1 = 0.f, a2 = 0.f;
#pragma unroll 8
    for (int d = 0; d < D_; ++d) { a1 += lq1[d] * lk1[d]; a2 += lq2[d] * lk2[d]; }
    const float lam = __expf(a1) - __expf(a2) + LAMBDA_INIT;

    const int m0 = qb * BM + wave * 16;   // first query row owned by this wave

    // ---- load Q tiles for both streams into A-layout fragments (bf16) ----
    // A layout (16-bit, 16x32): lanes 0-15 hold row=li with K = {0..7, 16..23}(+32*kk),
    // lanes 16-31 hold row=li with K = {8..15, 24..31}(+32*kk).
    Frag16 qf[2][2];  // [stream][kk]
    {
        const int row = m0 + li;
#pragma unroll
        for (int s = 0; s < 2; ++s) {
            const float* qrow = q_g + ((size_t)b * L_ + row) * DM_ + h * (2 * D_) + s * D_;
#pragma unroll
            for (int kk = 0; kk < 2; ++kk) {
                const int f0 = kk * 32 + hi * 8;        // -> elements 0..7
                const int f1 = kk * 32 + 16 + hi * 8;   // -> elements 8..15
#pragma unroll
                for (int i = 0; i < 8; ++i) qf[s][kk].u[i]     = f2bf(qrow[f0 + i]);
#pragma unroll
                for (int i = 0; i < 8; ++i) qf[s][kk].u[8 + i] = f2bf(qrow[f1 + i]);
            }
        }
    }

    // ---- online-softmax state: C-layout row = (vgpr j) + 8*hi ----
    float  mrow[2][8], lrow[2][8];
    Frag32 oacc[2][8];   // [stream][channel tile]; 128 accum VGPRs
#pragma unroll
    for (int s = 0; s < 2; ++s)
#pragma unroll
        for (int j = 0; j < 8; ++j) {
            mrow[s][j] = -1e30f;
            lrow[s][j] = 0.f;
#pragma unroll
            for (int f = 0; f < 8; ++f) oacc[s][f].f[j] = 0.f;
        }

    const float scale = 0.125f;  // 1/sqrt(64)

    // ---- software pipeline prologue: stage + convert block 0 ----
    stage_async(0);
    wait_async_all();
    __syncthreads();          // staging visible to all waves
    convert_tiles();
    __syncthreads();          // bf16 tiles ready; staging buffer free

    for (int kb = 0; kb <= qb; ++kb) {
        // overlap: stream next key block into staging while we compute on kb
        if (kb + 1 <= qb) stage_async(kb + 1);

        const bool diag = (kb == qb);

#pragma unroll
        for (int s = 0; s < 2; ++s) {
            // ---- S = Q K^T : 4 N-tiles x 2 K-steps of v_wmma_f32_16x16x32_bf16 ----
            Frag32 sc[4];
#pragma unroll
            for (int nt = 0; nt < 4; ++nt) {
                f32x8 c = {};
#pragma unroll
                for (int kk = 0; kk < 2; ++kk) {
                    Frag16 kb_frag;  // B layout: lane = key (li), K = feature pairs
                    const unsigned short* kp = &SK(s, nt * 16 + li, kk * 32 + hi * 8);
                    kb_frag.q[0] = *(const u32x4*)(kp);
                    kb_frag.q[1] = *(const u32x4*)(kp + 16);
                    c = __builtin_amdgcn_wmma_f32_16x16x32_bf16(
                            false, qf[s][kk].v, false, kb_frag.v, (short)0, c, false, false);
                }
                sc[nt].v = c;
#pragma unroll
                for (int j = 0; j < 8; ++j) sc[nt].f[j] *= scale;
                if (diag) {
                    const int col = kb * BN + nt * 16 + li;
#pragma unroll
                    for (int j = 0; j < 8; ++j)
                        if (col > m0 + j + 8 * hi) sc[nt].f[j] = NEGV;
                }
            }

            // ---- online softmax: row max / exp / row sum (16-lane butterfly) ----
            float alpha[8];
#pragma unroll
            for (int j = 0; j < 8; ++j) {
                float v = fmaxf(fmaxf(sc[0].f[j], sc[1].f[j]), fmaxf(sc[2].f[j], sc[3].f[j]));
                v = fmaxf(v, __shfl_xor(v, 1, 32));
                v = fmaxf(v, __shfl_xor(v, 2, 32));
                v = fmaxf(v, __shfl_xor(v, 4, 32));
                v = fmaxf(v, __shfl_xor(v, 8, 32));
                const float mn = fmaxf(mrow[s][j], v);
                alpha[j]  = __expf(mrow[s][j] - mn);
                mrow[s][j] = mn;
            }
            float rs[8];
#pragma unroll
            for (int j = 0; j < 8; ++j) rs[j] = 0.f;
#pragma unroll
            for (int nt = 0; nt < 4; ++nt)
#pragma unroll
                for (int j = 0; j < 8; ++j) {
                    const float p = __expf(sc[nt].f[j] - mrow[s][j]);
                    sc[nt].f[j] = p;
                    rs[j] += p;
                }
#pragma unroll
            for (int j = 0; j < 8; ++j) {
                float v = rs[j];
                v += __shfl_xor(v, 1, 32);
                v += __shfl_xor(v, 2, 32);
                v += __shfl_xor(v, 4, 32);
                v += __shfl_xor(v, 8, 32);
                lrow[s][j] = lrow[s][j] * alpha[j] + v;
#pragma unroll
                for (int f = 0; f < 8; ++f) oacc[s][f].f[j] *= alpha[j];
            }

            // ---- P: C-layout -> LDS -> A-layout (wave-private, no block barrier) ----
#pragma unroll
            for (int nt = 0; nt < 4; ++nt)
#pragma unroll
                for (int j = 0; j < 8; ++j)
                    SP(wave, s, j + 8 * hi, nt * 16 + li) = f2bf(sc[nt].f[j]);

            Frag16 pa[2];
#pragma unroll
            for (int kk = 0; kk < 2; ++kk) {
                const unsigned short* pp = &SP(wave, s, li, kk * 32 + hi * 8);
                pa[kk].q[0] = *(const u32x4*)(pp);        // cols base .. base+7
                pa[kk].q[1] = *(const u32x4*)(pp + 16);   // cols base+16 .. base+23
            }

            // ---- O += P V : 8 channel tiles x 2 K-steps ----
#pragma unroll
            for (int nt = 0; nt < 8; ++nt) {
                f32x8 c = oacc[s][nt].v;
#pragma unroll
                for (int kk = 0; kk < 2; ++kk) {
                    Frag16 vb;  // B layout: lane = channel (li), K = key pairs
                    const unsigned short* vp = &SVT(nt * 16 + li, kk * 32 + hi * 8);
                    vb.q[0] = *(const u32x4*)(vp);
                    vb.q[1] = *(const u32x4*)(vp + 16);
                    c = __builtin_amdgcn_wmma_f32_16x16x32_bf16(
                            false, pa[kk].v, false, vb.v, (short)0, c, false, false);
                }
                oacc[s][nt].v = c;
            }
        }

        // ---- pipeline epilogue for this block: publish kb+1's tiles ----
        if (kb + 1 <= qb) {
            wait_async_all();   // each wave drains its own ASYNCcnt
            __syncthreads();    // all compute on old tiles done; staging complete
            convert_tiles();    // overwrite sK/sVt with kb+1 (reads fK/fV)
            __syncthreads();    // tiles published; staging free for next issue
        }
    }

    // ---- epilogue: combine streams, per-row RMSNorm over 128 ch, store ----
    float wch[8];
#pragma unroll
    for (int f = 0; f < 8; ++f) wch[f] = w_g[f * 16 + li];

#pragma unroll
    for (int j = 0; j < 8; ++j) {
        const float i1 = 1.f / lrow[0][j];
        const float i2 = 1.f / lrow[1][j];
        float xr[8];
        float ssq = 0.f;
#pragma unroll
        for (int f = 0; f < 8; ++f) {
            const float t = oacc[0][f].f[j] * i1 - lam * (oacc[1][f].f[j] * i2);
            xr[f] = t;
            ssq += t * t;
        }
        ssq += __shfl_xor(ssq, 1, 32);
        ssq += __shfl_xor(ssq, 2, 32);
        ssq += __shfl_xor(ssq, 4, 32);
        ssq += __shfl_xor(ssq, 8, 32);
        const float rn = rsqrtf(ssq * (1.f / 128.f) + EPS_) * (1.f - LAMBDA_INIT);
        const int row = m0 + j + 8 * hi;
        float* orow = out + ((size_t)b * L_ + row) * DM_ + h * VD_;
#pragma unroll
        for (int f = 0; f < 8; ++f) orow[f * 16 + li] = xr[f] * rn * wch[f];
    }
}

extern "C" void kernel_launch(void* const* d_in, const int* in_sizes, int n_in,
                              void* d_out, int out_size, void* d_ws, size_t ws_size,
                              hipStream_t stream) {
    (void)in_sizes; (void)n_in; (void)out_size; (void)d_ws; (void)ws_size;
    const float* q   = (const float*)d_in[0];
    const float* k   = (const float*)d_in[1];
    const float* v   = (const float*)d_in[2];
    const float* lq1 = (const float*)d_in[3];
    const float* lk1 = (const float*)d_in[4];
    const float* lq2 = (const float*)d_in[5];
    const float* lk2 = (const float*)d_in[6];
    const float* w   = (const float*)d_in[7];
    float* o = (float*)d_out;

    dim3 grid(L_ / BM, B_ * H_);  // 32 x 16 = 512 workgroups
    dim3 block(NW * 32);          // 4 wave32 waves
    hipLaunchKernelGGL(diffattn_fa_kernel, grid, block, SMEM_BYTES, stream,
                       q, k, v, lq1, lk1, lq2, lk2, w, o);
}
